// CNNLSTMDecoder_1477468750098
// MI455X (gfx1250) — compile-verified, matrix operand order checked by reading, not run on hardware
//
#include <hip/hip_runtime.h>
#include <hip/hip_bf16.h>

// ---------------- problem dims ----------------
#define B_   128
#define CIN  256
#define T_   1024
#define CC   256      // conv out channels
#define HH   512      // LSTM hidden
#define G4   2048     // 4*HH
#define TPAD 1028     // T + 2*2 halo rows for k=5 conv
#define NWG_LSTM 32   // persistent workgroups in recurrence (HH/16)

typedef __bf16 bf16_t;
typedef __bf16 bf16x16 __attribute__((ext_vector_type(16)));
typedef __bf16 bf16x8  __attribute__((ext_vector_type(8)));
typedef float  f32x8   __attribute__((ext_vector_type(8)));
typedef int    i32x4   __attribute__((ext_vector_type(4)));

#define AS1 __attribute__((address_space(1)))
#define AS3 __attribute__((address_space(3)))

#if __has_builtin(__builtin_amdgcn_global_load_async_to_lds_b128) && \
    __has_builtin(__builtin_amdgcn_s_wait_asynccnt)
#define HAVE_ASYNC_LDS 1
#else
#define HAVE_ASYNC_LDS 0
#endif

// ---------------- helpers ----------------
__device__ __forceinline__ float sigf(float x) { return 1.0f / (1.0f + __expf(-x)); }

__device__ __forceinline__ void grid_barrier(volatile unsigned* cnt, unsigned target) {
    __syncthreads();
    if (threadIdx.x == 0) {
        __threadfence();
        atomicAdd((unsigned*)cnt, 1u);
        while (*cnt < target) { __builtin_amdgcn_s_sleep(2); }
        __threadfence();
    }
    __syncthreads();
}

// ---------------- init: zero barrier counter + h0 ----------------
__global__ void k_init(unsigned* cnt, bf16_t* hbf) {
    size_t i = (size_t)blockIdx.x * blockDim.x + threadIdx.x;
    if (i == 0) *cnt = 0u;
    if (i < (size_t)B_ * HH) hbf[i] = (bf16_t)0.0f;
}

// ---------------- weight packing ----------------
// conv_w [CC][CIN][5] (row-major) -> wpk [5][CC][CIN] bf16 (lane-contiguous K for WMMA B)
__global__ void k_pack_conv(const float* w, bf16_t* wpk) {
    size_t total = (size_t)5 * CC * CIN;
    for (size_t i = (size_t)blockIdx.x * blockDim.x + threadIdx.x; i < total;
         i += (size_t)gridDim.x * blockDim.x) {
        int tap = (int)(i / (CC * CIN));
        int rem = (int)(i % (CC * CIN));
        int co = rem / CIN, ci = rem % CIN;
        wpk[i] = (bf16_t)w[((size_t)co * CIN + ci) * 5 + tap];
    }
}

__global__ void k_cast_bf16(const float* src, bf16_t* dst, size_t n) {
    for (size_t i = (size_t)blockIdx.x * blockDim.x + threadIdx.x; i < n;
         i += (size_t)gridDim.x * blockDim.x)
        dst[i] = (bf16_t)src[i];
}

// ---------------- x: [B,CIN,T] f32 -> xT: [B,TPAD,CIN] bf16, halo rows zeroed ----------------
__global__ void k_transpose_x(const float* x, bf16_t* xT) {
    size_t total = (size_t)B_ * TPAD * CIN;
    for (size_t i = (size_t)blockIdx.x * blockDim.x + threadIdx.x; i < total;
         i += (size_t)gridDim.x * blockDim.x) {
        int ci = (int)(i % CIN);
        size_t r = i / CIN;
        int tp = (int)(r % TPAD);
        int b  = (int)(r / TPAD);
        float v = 0.0f;
        if (tp >= 2 && tp < T_ + 2)
            v = x[((size_t)b * CIN + ci) * T_ + (tp - 2)];
        xT[i] = (bf16_t)v;
    }
}

// ---------------- Conv1d as 5 accumulated WMMA GEMMs -> feats [T][B][CC] f32 ----------------
// grid: (B_, T_/128, CC/64), block: 256 (8 waves); wave w -> M rows [w*16, w*16+16)
__global__ void k_conv_wmma(const bf16_t* __restrict__ xT, const bf16_t* __restrict__ wpk,
                            float* __restrict__ feats) {
    const int b = blockIdx.x, tblk = blockIdx.y, nblk = blockIdx.z;
    const int wave = threadIdx.x >> 5, lane = threadIdx.x & 31;
    const int lr = lane & 15, lh = lane >> 4;
    const int m0 = tblk * 128 + wave * 16;   // t-row base (unpadded)
    const int n0 = nblk * 64;                // co base

    f32x8 acc[4] = {f32x8{0}, f32x8{0}, f32x8{0}, f32x8{0}};

    for (int tap = 0; tap < 5; ++tap) {
        // padded row index for A-tile row lr:  (m0 + lr) + tap
        const bf16_t* arow = xT + ((size_t)b * TPAD + (m0 + tap + lr)) * CIN;
        const bf16_t* wt   = wpk + (size_t)tap * CC * CIN;
        for (int kk = 0; kk < CIN; kk += 32) {
            bf16x16 a = *(const bf16x16*)(arow + kk + lh * 16);
#pragma unroll
            for (int nt = 0; nt < 4; ++nt) {
                bf16x16 bm = *(const bf16x16*)(wt + (size_t)(n0 + nt * 16 + lr) * CIN + kk + lh * 16);
                acc[nt] = __builtin_amdgcn_wmma_f32_16x16x32_bf16(
                    false, a, false, bm, (short)0, acc[nt], false, false);
            }
        }
    }
#pragma unroll
    for (int nt = 0; nt < 4; ++nt) {
#pragma unroll
        for (int r = 0; r < 8; ++r) {
            int t  = m0 + r + lh * 8;
            int co = n0 + nt * 16 + lr;
            feats[((size_t)t * B_ + b) * CC + co] = acc[nt][r];
        }
    }
}

// ---------------- BatchNorm batch stats per channel -> scale/shift ----------------
__global__ void k_bn_stats(const float* __restrict__ feats, const float* __restrict__ gamma,
                           const float* __restrict__ beta, float* scale, float* shift) {
    const int c = blockIdx.x;
    float s = 0.f, sq = 0.f;
    for (int i = threadIdx.x; i < T_ * B_; i += blockDim.x) {
        float v = feats[(size_t)i * CC + c];
        s += v; sq += v * v;
    }
    __shared__ float ss[256], s2[256];
    ss[threadIdx.x] = s; s2[threadIdx.x] = sq;
    __syncthreads();
    for (int o = 128; o > 0; o >>= 1) {
        if (threadIdx.x < o) { ss[threadIdx.x] += ss[threadIdx.x + o]; s2[threadIdx.x] += s2[threadIdx.x + o]; }
        __syncthreads();
    }
    if (threadIdx.x == 0) {
        const float inv_n = 1.0f / (float)(T_ * B_);
        float mean = ss[0] * inv_n;
        float var  = s2[0] * inv_n - mean * mean;
        float sc   = gamma[c] * rsqrtf(var + 1e-5f);
        scale[c] = sc;
        shift[c] = beta[c] - mean * sc;
    }
}

// ---------------- BN apply + ReLU + cast -> xs [T][B][CC] bf16 ----------------
__global__ void k_bn_apply(const float* __restrict__ feats, const float* __restrict__ scale,
                           const float* __restrict__ shift, bf16_t* __restrict__ xs) {
    size_t total = (size_t)T_ * B_ * CC;
    for (size_t i = (size_t)blockIdx.x * blockDim.x + threadIdx.x; i < total;
         i += (size_t)gridDim.x * blockDim.x) {
        int c = (int)(i % CC);
        float v = fmaxf(feats[i] * scale[c] + shift[c], 0.0f);
        xs[i] = (bf16_t)v;
    }
}

// ---------------- gx = xs @ w_ih^T + b_ih + b_hh  (WMMA) -> gx [T*B][G4] bf16 ----------------
// grid: (T*B/128, G4/64), block 256 (8 waves)
__global__ void k_gx_wmma(const bf16_t* __restrict__ xs, const bf16_t* __restrict__ wih,
                          const float* __restrict__ bih, const float* __restrict__ bhh,
                          bf16_t* __restrict__ gx) {
    const int wave = threadIdx.x >> 5, lane = threadIdx.x & 31;
    const int lr = lane & 15, lh = lane >> 4;
    const int m0 = blockIdx.x * 128 + wave * 16;
    const int n0 = blockIdx.y * 64;

    f32x8 acc[4] = {f32x8{0}, f32x8{0}, f32x8{0}, f32x8{0}};
    const bf16_t* arow = xs + (size_t)(m0 + lr) * CIN;
    for (int kk = 0; kk < CIN; kk += 32) {
        bf16x16 a = *(const bf16x16*)(arow + kk + lh * 16);
#pragma unroll
        for (int nt = 0; nt < 4; ++nt) {
            bf16x16 bm = *(const bf16x16*)(wih + (size_t)(n0 + nt * 16 + lr) * CIN + kk + lh * 16);
            acc[nt] = __builtin_amdgcn_wmma_f32_16x16x32_bf16(
                false, a, false, bm, (short)0, acc[nt], false, false);
        }
    }
#pragma unroll
    for (int nt = 0; nt < 4; ++nt) {
#pragma unroll
        for (int r = 0; r < 8; ++r) {
            int m = m0 + r + lh * 8;
            int n = n0 + nt * 16 + lr;
            gx[(size_t)m * G4 + n] = (bf16_t)(acc[nt][r] + bih[n] + bhh[n]);
        }
    }
}

// ---------------- persistent LSTM recurrence ----------------
// 32 WGs; WG w owns hidden columns [w*16, w*16+16). Wave i owns batch rows [i*16, i*16+16).
// The WG's 64KB w_hh slice is staged into LDS ONCE via async global->LDS copies,
// removing ~2GB of redundant L2 traffic from the serial critical path.
// c-state lives in VGPRs for all 1024 steps; h exchanged via global bf16 + grid barrier.
__global__ void k_lstm(const bf16_t* __restrict__ gx, const bf16_t* __restrict__ whh,
                       bf16_t* __restrict__ hbf, unsigned* cnt) {
    const int wg = blockIdx.x;
    const int wave = threadIdx.x >> 5, lane = threadIdx.x & 31;
    const int lr = lane & 15, lh = lane >> 4;
    const int row0 = wave * 16;   // batch rows
    const int j0 = wg * 16;       // hidden cols

    // 64 rows (4 gates x 16 cols) x 512 K of bf16 = 65536 B
    __shared__ bf16_t whh_lds[64 * HH];

    // ---- stage w_hh slice into LDS (async DMA path on gfx1250) ----
    {
        const int tid = threadIdx.x;
#pragma unroll 4
        for (int it = 0; it < 16; ++it) {
            int chunk = it * 256 + tid;      // 4096 chunks of 8 bf16 (16 B)
            int row   = chunk >> 6;          // 64 chunks per 512-elem row
            int kk8   = (chunk & 63) << 3;
            int g = row >> 4, c = row & 15;
            const bf16_t* src = whh + ((size_t)g * HH + j0 + c) * HH + kk8;
            bf16_t*       dst = &whh_lds[(size_t)row * HH + kk8];
#if HAVE_ASYNC_LDS
            __builtin_amdgcn_global_load_async_to_lds_b128(
                (AS1 i32x4*)src, (AS3 i32x4*)dst, 0, 0);
#else
            *(bf16x8*)dst = *(const bf16x8*)src;
#endif
        }
#if HAVE_ASYNC_LDS
        __builtin_amdgcn_s_wait_asynccnt(0);
#endif
        __syncthreads();
    }

    float cst[8] = {0, 0, 0, 0, 0, 0, 0, 0};
    float hnew[8];

    for (int t = 0; t < T_; ++t) {
        f32x8 acc[4] = {f32x8{0}, f32x8{0}, f32x8{0}, f32x8{0}};
        const bf16_t* arow = hbf + (size_t)(row0 + lr) * HH;
        for (int kk = 0; kk < HH; kk += 32) {
            bf16x16 a = *(const bf16x16*)(arow + kk + lh * 16);
#pragma unroll
            for (int g = 0; g < 4; ++g) {
                bf16x16 bm = *(const bf16x16*)(&whh_lds[(size_t)(g * 16 + lr) * HH + kk + lh * 16]);
                acc[g] = __builtin_amdgcn_wmma_f32_16x16x32_bf16(
                    false, a, false, bm, (short)0, acc[g], false, false);
            }
        }
#pragma unroll
        for (int r = 0; r < 8; ++r) {
            const int m = row0 + r + lh * 8;
            const int j = j0 + lr;
            const size_t gb = ((size_t)t * B_ + m) * G4;
            float iv = (float)gx[gb + 0 * HH + j] + acc[0][r];
            float fv = (float)gx[gb + 1 * HH + j] + acc[1][r];
            float gv = (float)gx[gb + 2 * HH + j] + acc[2][r];
            float ov = (float)gx[gb + 3 * HH + j] + acc[3][r];
            iv = sigf(iv); fv = sigf(fv); gv = tanhf(gv); ov = sigf(ov);
            cst[r] = fv * cst[r] + iv * gv;
            hnew[r] = ov * tanhf(cst[r]);
            if (t + 1 < T_)  // pull next step's gx slice toward L2/L0 (global_prefetch_b8)
                __builtin_prefetch(gx + gb + (size_t)B_ * G4 + j, 0, 3);
        }
        // all reads of h(t-1) done before anyone overwrites it
        grid_barrier(cnt, (unsigned)(2 * t + 1) * NWG_LSTM);
#pragma unroll
        for (int r = 0; r < 8; ++r)
            hbf[(size_t)(row0 + r + lh * 8) * HH + (j0 + lr)] = (bf16_t)hnew[r];
        // all writes of h(t) visible before next step reads
        grid_barrier(cnt, (unsigned)(2 * t + 2) * NWG_LSTM);
    }
}

// ---------------- head: Linear(512->64)+ReLU, Linear(64->2) ----------------
__global__ void k_head1(const bf16_t* __restrict__ hbf, const float* __restrict__ w1,
                        const float* __restrict__ b1, float* __restrict__ hid) {
    int idx = blockIdx.x * blockDim.x + threadIdx.x;
    if (idx >= B_ * 64) return;
    int b = idx >> 6, j = idx & 63;
    float s = b1[j];
    for (int k = 0; k < HH; ++k) s += (float)hbf[(size_t)b * HH + k] * w1[(size_t)j * HH + k];
    hid[idx] = fmaxf(s, 0.0f);
}

__global__ void k_head2(const float* __restrict__ hid, const float* __restrict__ w2,
                        const float* __restrict__ b2, float* __restrict__ out) {
    int idx = blockIdx.x * blockDim.x + threadIdx.x;
    if (idx >= B_ * 2) return;
    int b = idx >> 1, k = idx & 1;
    float s = b2[k];
    for (int j = 0; j < 64; ++j) s += hid[b * 64 + j] * w2[k * 64 + j];
    out[idx] = s;
}

// ---------------- launch ----------------
extern "C" void kernel_launch(void* const* d_in, const int* in_sizes, int n_in,
                              void* d_out, int out_size, void* d_ws, size_t ws_size,
                              hipStream_t stream) {
    const float* x       = (const float*)d_in[0];
    const float* conv_w  = (const float*)d_in[1];
    const float* bn_g    = (const float*)d_in[2];
    const float* bn_b    = (const float*)d_in[3];
    const float* w_ih    = (const float*)d_in[4];
    const float* w_hh    = (const float*)d_in[5];
    const float* b_ih    = (const float*)d_in[6];
    const float* b_hh    = (const float*)d_in[7];
    const float* w1      = (const float*)d_in[8];
    const float* b1      = (const float*)d_in[9];
    const float* w2      = (const float*)d_in[10];
    const float* b2      = (const float*)d_in[11];
    float* out = (float*)d_out;

    // workspace carve-out (256B aligned)
    size_t off = 0;
    auto alloc = [&](size_t bytes) -> void* {
        void* p = (char*)d_ws + off;
        off += (bytes + 255) & ~(size_t)255;
        return p;
    };
    unsigned* cnt   = (unsigned*)alloc(256);
    bf16_t* hbf     = (bf16_t*)alloc((size_t)B_ * HH * sizeof(bf16_t));
    bf16_t* xT      = (bf16_t*)alloc((size_t)B_ * TPAD * CIN * sizeof(bf16_t));
    bf16_t* wpk     = (bf16_t*)alloc((size_t)5 * CC * CIN * sizeof(bf16_t));
    bf16_t* wih_bf  = (bf16_t*)alloc((size_t)G4 * CIN * sizeof(bf16_t));
    bf16_t* whh_bf  = (bf16_t*)alloc((size_t)G4 * HH * sizeof(bf16_t));
    float*  scale   = (float*)alloc(CC * sizeof(float));
    float*  shift   = (float*)alloc(CC * sizeof(float));
    float*  feats   = (float*)alloc((size_t)T_ * B_ * CC * sizeof(float));
    bf16_t* xs      = (bf16_t*)alloc((size_t)T_ * B_ * CC * sizeof(bf16_t));
    bf16_t* gx      = (bf16_t*)alloc((size_t)T_ * B_ * G4 * sizeof(bf16_t));
    float*  hid     = (float*)alloc((size_t)B_ * 64 * sizeof(float));
    (void)ws_size; (void)n_in; (void)in_sizes; (void)out_size;

    k_init<<<(B_ * HH + 255) / 256, 256, 0, stream>>>(cnt, hbf);
    k_pack_conv<<<1280, 256, 0, stream>>>(conv_w, wpk);
    k_cast_bf16<<<2048, 256, 0, stream>>>(w_ih, wih_bf, (size_t)G4 * CIN);
    k_cast_bf16<<<4096, 256, 0, stream>>>(w_hh, whh_bf, (size_t)G4 * HH);
    k_transpose_x<<<8192, 256, 0, stream>>>(x, xT);

    k_conv_wmma<<<dim3(B_, T_ / 128, CC / 64), 256, 0, stream>>>(xT, wpk, feats);
    k_bn_stats<<<CC, 256, 0, stream>>>(feats, bn_g, bn_b, scale, shift);
    k_bn_apply<<<8192, 256, 0, stream>>>(feats, scale, shift, xs);

    k_gx_wmma<<<dim3((T_ * B_) / 128, G4 / 64), 256, 0, stream>>>(xs, wih_bf, b_ih, b_hh, gx);

    k_lstm<<<NWG_LSTM, 256, 0, stream>>>(gx, whh_bf, hbf, cnt);

    k_head1<<<(B_ * 64 + 255) / 256, 256, 0, stream>>>(hbf, w1, b1, hid);
    k_head2<<<1, 256, 0, stream>>>(hid, w2, b2, out);
}
